// MultiHeadEncoder_37486474559884
// MI455X (gfx1250) — matrix operand order repaired and checked by reading, not compiled, not run on hardware
//
#include <hip/hip_runtime.h>
#include <hip/hip_bf16.h>
#include <stdint.h>

// ---------------------------------------------------------------------------
// MultiHeadEncoder for MI455X (gfx1250, wave32, WMMA bf16 -> f32 accumulate)
//
// Convs are implicit GEMMs: M = B*Do*Ho*Wo voxels, N = Cout, K = taps*Cin.
// Activations channels-last bf16; weights pre-transformed to [K, N] bf16.
// 128-thread block (4 waves): block tile 128(M) x 32(N); each wave owns a
// 32x32 C tile = four v_wmma_f32_16x16x32_bf16 per K-step (2 A-frags x
// 2 B-frags). Cin is always a power of two -> shift/mask K decomposition +
// LDS tap LUT (no integer division in the hot loop). The weight tile is
// staged with global_load_async_to_lds_b128 + s_wait_asynccnt.
// ---------------------------------------------------------------------------

typedef __attribute__((ext_vector_type(16))) __bf16 v16bf;
typedef __attribute__((ext_vector_type(8)))  float  v8f;
typedef __attribute__((ext_vector_type(4)))  int    v4i;

union BFrag { v16bf v; unsigned short u[16]; };

#if defined(__AMDGCN__) &&                                             \
    __has_builtin(__builtin_amdgcn_global_load_async_to_lds_b128) &&   \
    __has_builtin(__builtin_amdgcn_s_wait_asynccnt)
#define USE_ASYNC_LDS 1
#define GAS1_V4I(p) ((__attribute__((address_space(1))) v4i*)(uintptr_t)(p))
#define LAS3_V4I(p) ((__attribute__((address_space(3))) v4i*)(uintptr_t)(p))
#else
#define USE_ASYNC_LDS 0
#endif

__device__ __forceinline__ unsigned short f2bf(float f) {
    unsigned int u = __float_as_uint(f);
    unsigned int r = u + 0x7FFFu + ((u >> 16) & 1u);   // round-to-nearest-even
    return (unsigned short)(r >> 16);
}

// ------------------------------ conv GEMM ----------------------------------
__global__ __launch_bounds__(128)
void conv_wmma_kernel(const unsigned short* __restrict__ in,  // bf16 NDHWC
                      const unsigned short* __restrict__ wk,  // bf16 [Ktot, Cout]
                      float* __restrict__ out,                // f32  [M, Cout]
                      int Din, int Hin, int Win, int lc,      // lc = log2(Cin)
                      int Do, int Ho, int Wo, int Cout, int Mtot,
                      int KD, int KH, int KW,
                      int sd, int sh, int sw,
                      int pd, int ph, int pw)
{
    __shared__ unsigned short lA[128][40];  // 128 rows x 32 K (+8 pad)
    __shared__ unsigned short lB[32][40];   // 32 K    x 32 N (+8 pad)
    __shared__ int            lut[32];      // tap -> packed (kd,kh,kw)

    const int tid  = threadIdx.x;
    const int wv   = tid >> 5;
    const int lane = tid & 31;

    const int taps = KD * KH * KW;
    const int Ktot = taps << lc;
    const int cm   = (1 << lc) - 1;

    // element strides in the NDHWC input
    const int sW = 1 << lc;
    const int sH = Win * sW;
    const int sD = Hin * sH;

    if (tid < taps) {
        const int kw_ = tid % KW;
        const int tt  = tid / KW;
        lut[tid] = ((tt / KH) << 16) | ((tt % KH) << 8) | kw_;
    }

    const int mtile = blockIdx.x * 128;
    const int n0    = blockIdx.y * 32;
    const int HWo   = Ho * Wo;
    const int DHWo  = Do * HWo;

    // One voxel row per thread for A staging.
    const int am      = mtile + tid;
    const bool mvalid = (am < Mtot);
    int bd = 0, bh = 0, bw = 0;
    size_t rowb = 0;
    if (mvalid) {
        int r = am;
        const int vb = r / DHWo; r -= vb * DHWo;
        const int od = r / HWo;  r -= od * HWo;
        const int oh = r / Wo;
        const int ow = r - oh * Wo;
        bd = od * sd - pd;
        bh = oh * sh - ph;
        bw = ow * sw - pw;
        rowb = (size_t)vb * Din * sD;
    }
    __syncthreads();   // lut visible

    v8f acc[2][2] = {{{}, {}}, {{}, {}}};

    for (int k0 = 0; k0 < Ktot; k0 += 32) {
        // ---- stage A tile: im2col gather, shift/mask + LDS tap LUT ----
        #pragma unroll 8
        for (int i = 0; i < 32; ++i) {
            const int k = k0 + i;
            unsigned short val = 0;
            if (mvalid && k < Ktot) {
                const int pk = lut[k >> lc];
                const int ci = k & cm;
                const int id = bd + (pk >> 16);
                const int ih = bh + ((pk >> 8) & 255);
                const int iw = bw + (pk & 255);
                if ((unsigned)id < (unsigned)Din && (unsigned)ih < (unsigned)Hin &&
                    (unsigned)iw < (unsigned)Win)
                    val = in[rowb + (size_t)id * sD + ih * sH + iw * sW + ci];
            }
            lA[tid][i] = val;
        }

        // ---- stage B tile (weights [K,N]): 8 bf16 = 16 bytes per thread ----
        // NOTE: reads past row end for Cout<32 stay inside the weight buffer;
        // those columns are discarded by the guarded store.
        {
            const int idx = tid * 8;
            const int kk  = idx >> 5;           // 0..31
            const int nn  = idx & 31;           // 0,8,16,24
            const int k   = k0 + kk;
            if (k < Ktot) {
#if USE_ASYNC_LDS
                __builtin_amdgcn_global_load_async_to_lds_b128(
                    GAS1_V4I(wk + (size_t)k * Cout + n0 + nn),
                    LAS3_V4I(&lB[kk][nn]), 0, 0);
#else
                const unsigned short* src = wk + (size_t)k * Cout + n0 + nn;
                #pragma unroll
                for (int j = 0; j < 8; ++j) lB[kk][nn + j] = src[j];
#endif
            } else {
                #pragma unroll
                for (int j = 0; j < 8; ++j) lB[kk][nn + j] = 0;
            }
        }

        // prefetch next weight tile while this one is consumed
        __builtin_prefetch(wk + (size_t)(k0 + 32) * Cout + n0 + (tid & 31), 0, 1);

#if USE_ASYNC_LDS
        __builtin_amdgcn_s_wait_asynccnt(0);
#endif
        __syncthreads();

        // ---- fragments: A and B both use contiguous k = 16*(lane/16) + i ----
        BFrag b0, b1;
        {
            const int nn = lane & 15;
            const int kb = (lane < 16) ? 0 : 16;
            #pragma unroll
            for (int i = 0; i < 16; ++i) {
                b0.u[i] = lB[kb + i][nn];
                b1.u[i] = lB[kb + i][16 + nn];
            }
        }
        BFrag a0, a1;
        {
            const int kb = (lane < 16) ? 0 : 16;
            const unsigned short* ap0 = &lA[wv * 32 + (lane & 15)][0];
            const unsigned short* ap1 = ap0 + 16 * 40;
            #pragma unroll
            for (int i = 0; i < 8; ++i) {
                a0.u[i]     = ap0[kb + i];
                a0.u[8 + i] = ap0[kb + 8 + i];
                a1.u[i]     = ap1[kb + i];
                a1.u[8 + i] = ap1[kb + 8 + i];
            }
        }

        acc[0][0] = __builtin_amdgcn_wmma_f32_16x16x32_bf16(
                        false, a0.v, false, b0.v, (short)0, acc[0][0], false, false);
        acc[0][1] = __builtin_amdgcn_wmma_f32_16x16x32_bf16(
                        false, a0.v, false, b1.v, (short)0, acc[0][1], false, false);
        acc[1][0] = __builtin_amdgcn_wmma_f32_16x16x32_bf16(
                        false, a1.v, false, b0.v, (short)0, acc[1][0], false, false);
        acc[1][1] = __builtin_amdgcn_wmma_f32_16x16x32_bf16(
                        false, a1.v, false, b1.v, (short)0, acc[1][1], false, false);

        __syncthreads();
    }

    // ---- store C (f32 16x16 layout: VGPR r -> M = r, lanes>=16 -> +8) ----
    {
        const int nn = lane & 15;
        #pragma unroll
        for (int sub = 0; sub < 2; ++sub) {
            const int mbase = mtile + wv * 32 + sub * 16 + ((lane < 16) ? 0 : 8);
            #pragma unroll
            for (int nsub = 0; nsub < 2; ++nsub) {
                const int col = n0 + nsub * 16 + nn;
                if (col >= Cout) continue;
                #pragma unroll
                for (int r = 0; r < 8; ++r) {
                    const int m = mbase + r;
                    if (m < Mtot) out[(size_t)m * Cout + col] = acc[sub][nsub][r];
                }
            }
        }
    }
}

// ------------------------- weight re-layout --------------------------------
// fp32 [Cout, Cin, KD, KH, KW]  ->  bf16 [taps*Cin, Cout]
__global__ void wxform_kernel(const float* __restrict__ w, unsigned short* __restrict__ o,
                              int Cout, int Cin, int KD, int KH, int KW, int total)
{
    const int i = blockIdx.x * blockDim.x + threadIdx.x;
    if (i >= total) return;
    const int co = i % Cout;
    const int k  = i / Cout;
    const int ci = k % Cin;
    const int t  = k / Cin;
    const int kw_ = t % KW;
    const int tt  = t / KW;
    const int kh_ = tt % KH;
    const int kd_ = tt / KH;
    const size_t src = ((((size_t)co * Cin + ci) * KD + kd_) * KH + kh_) * KW + kw_;
    o[i] = f2bf(w[src]);
}

// -------------------------- input prep -------------------------------------
// x NCDHW fp32 [2,4,16,128,128] * mask -> bf16 NDHWC [524288, 4]
__global__ void prep_input_kernel(const float* __restrict__ x, const float* __restrict__ m,
                                  unsigned short* __restrict__ o)
{
    const int v = blockIdx.x * blockDim.x + threadIdx.x;
    if (v >= 524288) return;
    const int b  = v >> 18;
    const int sp = v & 262143;
    const float mv = m[v];
    #pragma unroll
    for (int ci = 0; ci < 4; ++ci)
        o[(size_t)v * 4 + ci] = f2bf(x[((size_t)(b * 4 + ci)) * 262144 + sp] * mv);
}

// ----------------------- sparse mask propagation ---------------------------
__global__ void mask_pool_kernel(const float* __restrict__ mi, float* __restrict__ mo,
                                 int Din, int Hin, int Win, int Do, int Ho, int Wo,
                                 int KD, int KH, int KW, int sd, int sh, int sw,
                                 int pd, int ph, int pw)
{
    const int HWo = Ho * Wo, DHWo = Do * HWo, Mtot = 2 * DHWo;
    const int i = blockIdx.x * blockDim.x + threadIdx.x;
    if (i >= Mtot) return;
    int r = i;
    const int b  = r / DHWo; r -= b * DHWo;
    const int od = r / HWo;  r -= od * HWo;
    const int oh = r / Wo;
    const int ow = r - oh * Wo;
    float acc = 0.f;
    for (int kd = 0; kd < KD; ++kd) {
        const int id = od * sd - pd + kd;
        if ((unsigned)id >= (unsigned)Din) continue;
        for (int kh = 0; kh < KH; ++kh) {
            const int ih = oh * sh - ph + kh;
            if ((unsigned)ih >= (unsigned)Hin) continue;
            for (int kw = 0; kw < KW; ++kw) {
                const int iw = ow * sw - pw + kw;
                if ((unsigned)iw >= (unsigned)Win) continue;
                acc += mi[(((size_t)b * Din + id) * Hin + ih) * (size_t)Win + iw];
            }
        }
    }
    mo[i] = (acc > 0.f) ? 1.f : 0.f;
}

// ------------------------------ misc ---------------------------------------
__global__ void zero_kernel(float* __restrict__ p, int n)
{
    for (int i = threadIdx.x + blockIdx.x * blockDim.x; i < n; i += blockDim.x * gridDim.x)
        p[i] = 0.f;
}

// ------------------------- masked BN statistics ----------------------------
// stats layout: [0,C) sum(x*m), [C,2C) sum(x^2*m), [2C] cnt = sum(m)
__global__ __launch_bounds__(256)
void bn_stats_kernel(const float* __restrict__ x, const float* __restrict__ mask,
                     float* __restrict__ stats, int Nvox, int C)
{
    extern __shared__ float sm[];
    for (int i = threadIdx.x; i < 2 * C + 1; i += blockDim.x) sm[i] = 0.f;
    __syncthreads();
    const int v = blockIdx.x * blockDim.x + threadIdx.x;
    if (v < Nvox) {
        const float m = mask[v];
        if (m != 0.f) {
            const float* row = x + (size_t)v * C;
            for (int c = 0; c < C; ++c) {
                const float val = row[c];
                atomicAdd(&sm[c],     val * m);
                atomicAdd(&sm[C + c], val * val * m);
            }
            atomicAdd(&sm[2 * C], m);
        }
    }
    __syncthreads();
    for (int i = threadIdx.x; i < 2 * C + 1; i += blockDim.x) {
        const float p = sm[i];
        if (p != 0.f) atomicAdd(&stats[i], p);
    }
}

// ---------------- BN normalize + ReLU + mask -> bf16 NDHWC -----------------
__global__ void bn_apply_kernel(const float* __restrict__ x, const float* __restrict__ mask,
                                const float* __restrict__ stats,
                                const float* __restrict__ g, const float* __restrict__ bb,
                                unsigned short* __restrict__ out,
                                int Nvox, int C, int outStride, int outOff)
{
    const int i = blockIdx.x * blockDim.x + threadIdx.x;
    if (i >= Nvox * C) return;
    const int v = i / C;
    const int c = i - v * C;
    const float cnt  = stats[2 * C];
    const float mean = stats[c] / cnt;
    const float var  = stats[C + c] / cnt - mean * mean;
    const float inv  = rsqrtf(var + 1e-3f);
    float y = g[c] * (x[(size_t)v * C + c] - mean) * inv + bb[c];
    y = fmaxf(y, 0.f) * mask[v];
    out[(size_t)v * outStride + outOff + c] = f2bf(y);
}

// ---------------- final BN + ReLU + mask -> f32 NCDHW into d_out -----------
__global__ void bn_apply_final_kernel(const float* __restrict__ x, const float* __restrict__ mask,
                                      const float* __restrict__ stats,
                                      const float* __restrict__ g, const float* __restrict__ bb,
                                      float* __restrict__ out)   // [2,64,1,32,32]
{
    const int C = 64;
    const int i = blockIdx.x * blockDim.x + threadIdx.x;
    if (i >= 2048 * C) return;
    const int v = i / C;
    const int c = i - v * C;
    const int b  = v >> 10;
    const int sp = v & 1023;
    const float cnt  = stats[2 * C];
    const float mean = stats[c] / cnt;
    const float var  = stats[C + c] / cnt - mean * mean;
    const float inv  = rsqrtf(var + 1e-3f);
    float y = g[c] * (x[(size_t)v * C + c] - mean) * inv + bb[c];
    y = fmaxf(y, 0.f) * mask[v];
    out[((size_t)(b * C + c)) * 1024 + sp] = y;
}

__global__ void copy_mask_kernel(const float* __restrict__ m, float* __restrict__ out)
{
    const int i = blockIdx.x * blockDim.x + threadIdx.x;
    if (i < 2048) out[i] = m[i];
}

// ===========================================================================
extern "C" void kernel_launch(void* const* d_in, const int* in_sizes, int n_in,
                              void* d_out, int out_size, void* d_ws, size_t ws_size,
                              hipStream_t stream)
{
    (void)in_sizes; (void)n_in; (void)out_size; (void)ws_size;

    const float* x  = (const float*)d_in[0];
    const float* m0 = (const float*)d_in[1];
    // params flattened as a pytree (dict keys alphabetical):
    // heads[h].blocks[blk]{b,g,w} -> red{b,g,w} -> shared{b,g,w}
    auto hp = [&](int h, int blk, int which) -> const float* {
        return (const float*)d_in[2 + h * 24 + blk * 3 + which];   // 0=b,1=g,2=w
    };
    const float* red_b = (const float*)d_in[50];
    const float* red_g = (const float*)d_in[51];
    const float* red_w = (const float*)d_in[52];
    const float* sh_b  = (const float*)d_in[53];
    const float* sh_g  = (const float*)d_in[54];
    const float* sh_w  = (const float*)d_in[55];

    // ---------------- workspace carve-up ----------------
    char* wsb = (char*)d_ws;
    size_t off = 0;
    auto carve = [&](size_t bytes) -> void* {
        void* p = wsb + off;
        off += (bytes + 255) & ~(size_t)255;
        return p;
    };
    unsigned short* ACT0 = (unsigned short*)carve(16777216ull * 2); // 32ch @ full res
    unsigned short* ACT1 = (unsigned short*)carve(16777216ull * 2);
    unsigned short* XS   = (unsigned short*)carve(8388608ull * 2);  // shared stage, 16ch
    unsigned short* X4   = (unsigned short*)carve(2097152ull * 2);  // masked input, 4ch
    float*          CONV = (float*)carve(16777216ull * 4);          // conv out f32
    unsigned short* CAT  = (unsigned short*)carve(524288ull * 2);   // [2048, 256]
    float* M1 = (float*)carve(65536ull * 4);
    float* M2 = (float*)carve(6144ull * 4);
    float* M3 = (float*)carve(2048ull * 4);
    unsigned short* WBUF = (unsigned short*)carve(262144ull);       // bf16 weights [K,N]
    float* STATS = (float*)carve(4096);

    // ---------------- helpers ----------------
    auto conv = [&](const unsigned short* in, float* out_,
                    int Din, int Hin, int Win, int Cin,
                    int Do, int Ho, int Wo, int Cout,
                    int KD, int KH, int KW, int sd, int sh, int sw,
                    int pd, int ph, int pw) {
        const int Mtot = 2 * Do * Ho * Wo;
        const int lc   = __builtin_ctz((unsigned)Cin);
        dim3 grid((Mtot + 127) / 128, (Cout + 31) / 32);
        conv_wmma_kernel<<<grid, 128, 0, stream>>>(in, WBUF, out_,
            Din, Hin, Win, lc, Do, Ho, Wo, Cout, Mtot,
            KD, KH, KW, sd, sh, sw, pd, ph, pw);
    };
    auto wxf = [&](const float* w, int Cout, int Cin, int KD, int KH, int KW) {
        const int tot = Cout * Cin * KD * KH * KW;
        wxform_kernel<<<(tot + 255) / 256, 256, 0, stream>>>(w, WBUF, Cout, Cin, KD, KH, KW, tot);
    };
    auto bn = [&](const float* cin_, const float* mask, const float* g, const float* bb,
                  unsigned short* aout, int Nvox, int C, int stride, int coff) {
        zero_kernel<<<1, 256, 0, stream>>>(STATS, 2 * C + 1);
        bn_stats_kernel<<<(Nvox + 255) / 256, 256, (2 * C + 1) * sizeof(float), stream>>>(
            cin_, mask, STATS, Nvox, C);
        const int tot = Nvox * C;
        bn_apply_kernel<<<(tot + 255) / 256, 256, 0, stream>>>(
            cin_, mask, STATS, g, bb, aout, Nvox, C, stride, coff);
    };

    // ---------------- masks for strided (sp) convs ----------------
    mask_pool_kernel<<<(2 * 8 * 64 * 64 + 255) / 256, 256, 0, stream>>>(
        m0, M1, 16, 128, 128, 8, 64, 64, 3, 3, 3, 2, 2, 2, 1, 1, 1);
    mask_pool_kernel<<<(2 * 3 * 32 * 32 + 255) / 256, 256, 0, stream>>>(
        M1, M2, 8, 64, 64, 3, 32, 32, 3, 3, 3, 2, 2, 2, 0, 1, 1);
    mask_pool_kernel<<<(2 * 1 * 32 * 32 + 255) / 256, 256, 0, stream>>>(
        M2, M3, 3, 32, 32, 1, 32, 32, 3, 1, 1, 2, 1, 1, 0, 0, 0);

    // ---------------- shared stem: 4 -> 16 ----------------
    prep_input_kernel<<<(524288 + 255) / 256, 256, 0, stream>>>(x, m0, X4);
    wxf(sh_w, 16, 4, 3, 3, 3);
    conv(X4, CONV, 16, 128, 128, 4, 16, 128, 128, 16, 3, 3, 3, 1, 1, 1, 1, 1, 1);
    bn(CONV, m0, sh_g, sh_b, XS, 524288, 16, 16, 0);

    // ---------------- per-head block specs ----------------
    struct Spec {
        int cin, cout, kd, kh, kw, sd, sh, sw, pd, ph, pw;
        int Din, Hin, Win, Do, Ho, Wo;
        const float* maskOut;
    };
    const Spec specs[8] = {
        {16,  32, 3,3,3, 1,1,1, 1,1,1, 16,128,128, 16,128,128, m0},
        {32,  64, 3,3,3, 2,2,2, 1,1,1, 16,128,128,  8, 64, 64, M1},
        {64,  64, 3,3,3, 1,1,1, 1,1,1,  8, 64, 64,  8, 64, 64, M1},
        {64,  64, 3,3,3, 1,1,1, 1,1,1,  8, 64, 64,  8, 64, 64, M1},
        {64,  64, 3,3,3, 2,2,2, 0,1,1,  8, 64, 64,  3, 32, 32, M2},
        {64,  64, 3,3,3, 1,1,1, 1,1,1,  3, 32, 32,  3, 32, 32, M2},
        {64,  64, 3,3,3, 1,1,1, 1,1,1,  3, 32, 32,  3, 32, 32, M2},
        {64, 128, 3,1,1, 2,1,1, 0,0,0,  3, 32, 32,  1, 32, 32, M3},
    };

    for (int h = 0; h < 2; ++h) {
        const unsigned short* cur = XS;
        for (int blk = 0; blk < 8; ++blk) {
            const Spec& s = specs[blk];
            wxf(hp(h, blk, 2), s.cout, s.cin, s.kd, s.kh, s.kw);
            conv(cur, CONV, s.Din, s.Hin, s.Win, s.cin, s.Do, s.Ho, s.Wo, s.cout,
                 s.kd, s.kh, s.kw, s.sd, s.sh, s.sw, s.pd, s.ph, s.pw);
            const int Nv = 2 * s.Do * s.Ho * s.Wo;
            if (blk == 7) {
                // write head output into concat buffer [2048, 256], col offset h*128
                bn(CONV, s.maskOut, hp(h, blk, 1), hp(h, blk, 0), CAT, Nv, s.cout, 256, h * 128);
            } else {
                unsigned short* nxt = (blk & 1) ? ACT1 : ACT0;
                bn(CONV, s.maskOut, hp(h, blk, 1), hp(h, blk, 0), nxt, Nv, s.cout, s.cout, 0);
                cur = nxt;
            }
        }
    }

    // ---------------- 1x1 reduction 256 -> 64 (GEMM via same conv path) ----
    wxf(red_w, 64, 256, 1, 1, 1);
    conv(CAT, CONV, 1, 32, 32, 256, 1, 32, 32, 64, 1, 1, 1, 1, 1, 1, 0, 0, 0);
    zero_kernel<<<1, 256, 0, stream>>>(STATS, 2 * 64 + 1);
    bn_stats_kernel<<<(2048 + 255) / 256, 256, (2 * 64 + 1) * sizeof(float), stream>>>(
        CONV, M3, STATS, 2048, 64);
    bn_apply_final_kernel<<<(2048 * 64 + 255) / 256, 256, 0, stream>>>(
        CONV, M3, STATS, red_g, red_b, (float*)d_out);
    copy_mask_kernel<<<(2048 + 255) / 256, 256, 0, stream>>>(M3, (float*)d_out + 131072);
}
